// ScaledDotProductAttention_55241869361696
// MI455X (gfx1250) — matrix operand order repaired and checked by reading, not compiled, not run on hardware
//
#include <hip/hip_runtime.h>

// ---------------- types ----------------
typedef __bf16 bf16_t;
typedef __attribute__((ext_vector_type(8)))  __bf16       v8bf;
typedef __attribute__((ext_vector_type(16))) __bf16       v16bf;
typedef __attribute__((ext_vector_type(8)))  float        v8f;
typedef __attribute__((ext_vector_type(4)))  float        v4f;
typedef __attribute__((ext_vector_type(4)))  unsigned int v4u;
typedef __attribute__((ext_vector_type(8)))  int          v8i;
typedef __attribute__((ext_vector_type(4)))  int          v4i;

#if __has_builtin(__builtin_amdgcn_tensor_load_to_lds) && __has_builtin(__builtin_amdgcn_s_wait_tensorcnt)
#define USE_TDM 1
#else
#define USE_TDM 0
#endif

#define ROWPAD 40  // bf16 elems per LDS row: 32 data + 8 pad (80B: stride-20 banks -> conflict-free)

// ---------------- fragment helpers ----------------
__device__ __forceinline__ v8bf ld8bf(const bf16_t* p) { return *(const v8bf*)p; }

__device__ __forceinline__ v16bf combine16(v8bf c0, v8bf c1) {
  v16bf r;
#pragma unroll
  for (int i = 0; i < 8; ++i) { r[i] = c0[i]; r[i + 8] = c1[i]; }
  return r;
}

__device__ __forceinline__ v8bf cvt8(const float* p) {
  v4f a = *(const v4f*)p;
  v4f b = *(const v4f*)(p + 4);
  v8bf r;
#pragma unroll
  for (int i = 0; i < 4; ++i) { r[i] = (__bf16)a[i]; r[i + 4] = (__bf16)b[i]; }
  return r;
}

// A fragment (16x32): lane holds row M=lane&15; contiguous 16 K values
__device__ __forceinline__ v16bf loadA_bf (const bf16_t* rk) { return combine16(ld8bf(rk), ld8bf(rk + 8)); }
__device__ __forceinline__ v16bf loadA_f32(const float*  rk) { return combine16(cvt8(rk),  cvt8(rk + 8)); }
// B fragment (32x16): lane holds col N=lane&15; contiguous 16 K values
__device__ __forceinline__ v16bf loadB_bf (const bf16_t* rk) { return combine16(ld8bf(rk), ld8bf(rk + 8)); }
__device__ __forceinline__ v16bf loadB_f32(const float*  rk) { return combine16(cvt8(rk),  cvt8(rk + 8)); }

__device__ __forceinline__ v8f wmma_bf16(v16bf a, v16bf b, v8f c) {
  return __builtin_amdgcn_wmma_f32_16x16x32_bf16(false, a, false, b, (short)0, c, false, false);
}

#if USE_TDM
// Issue a TDM copy of a (128 rows x 32 cols) bf16 tile into LDS.
// D# per CDNA5 ISA ch.8: 2D tensor, data_size=2B, LDS pad 4 dwords per 16 dwords (80B rows).
// Toolchain uses the 6-arg builtin form: (g0, g1, g2, g3, g4, cpol).
__device__ __forceinline__ void tdm_load_tile(const bf16_t* gsrc, unsigned lds_off,
                                              int dim0, int dim1, int stride) {
  unsigned long long ga = (unsigned long long)(size_t)gsrc;
  v4u g0;
  g0[0] = 1u;                                                  // count=1, user mode
  g0[1] = lds_off;                                             // lds_addr (bytes)
  g0[2] = (unsigned)ga;                                        // global_addr[31:0]
  g0[3] = (unsigned)((ga >> 32) & 0x01FFFFFFull) | (2u << 30); // global_addr[56:32], type=2
  v8i g1;
  g1[0] = (int)((1u << 16)    // data_size = 2 bytes
              | (1u << 20)    // pad_enable
              | (3u << 22)    // pad_interval: 16 dwords (64B)
              | (3u << 25));  // pad_amount: 4 dwords (16B)
  g1[1] = (int)((unsigned)(dim0 & 0xFFFF) << 16);                            // tensor_dim0 lo
  g1[2] = (int)(((unsigned)dim0 >> 16) | ((unsigned)(dim1 & 0xFFFF) << 16)); // dim0 hi | dim1 lo
  g1[3] = (int)(((unsigned)dim1 >> 16) | (32u << 16));                       // dim1 hi | tile_dim0=32
  g1[4] = 128;                                                               // tile_dim1=128
  g1[5] = stride;                                                            // tensor_dim0_stride lo
  g1[6] = 0;
  g1[7] = 0;
  v4i z4 = {0, 0, 0, 0};
  v8i z8 = {0, 0, 0, 0, 0, 0, 0, 0};
  __builtin_amdgcn_tensor_load_to_lds(g0, g1, z4, z4, z8, 0);
}
#endif

// Fallback cooperative staging: 128x32 bf16 tile -> LDS (padded rows)
__device__ __forceinline__ void stage_tile(const bf16_t* __restrict__ g, bf16_t* dst,
                                           int K, int tid) {
#pragma unroll
  for (int c = tid; c < 512; c += 256) {
    const int row = c >> 2, col = (c & 3) << 3;
    *(v8bf*)(dst + row * ROWPAD + col) = *(const v8bf*)(g + (size_t)row * K + col);
  }
}

// 8 WMMAs on the staged 128x32 slabs (per-wave 32x64 tile)
__device__ __forceinline__ void compute_tiles(const bf16_t* Ab, const bf16_t* Bb,
                                              int ml0, int nl0, int lm, int lh,
                                              v8f acc[2][4]) {
  const int offA = lh * 16, offB = lh * 16;
  v16bf a0 = loadA_bf(Ab + (ml0 + lm) * ROWPAD + offA);
  v16bf a1 = loadA_bf(Ab + (ml0 + 16 + lm) * ROWPAD + offA);
#pragma unroll
  for (int nt = 0; nt < 4; ++nt) {
    v16bf b = loadB_bf(Bb + (nl0 + nt * 16 + lm) * ROWPAD + offB);
    acc[0][nt] = wmma_bf16(a0, b, acc[0][nt]);
    acc[1][nt] = wmma_bf16(a1, b, acc[1][nt]);
  }
}

// ================= projection: P[m,f] = sum_e X[m,e]*W[f,e] + b[f] =================
__global__ __launch_bounds__(256) void proj_kernel(
    const float* __restrict__ X, const float* __restrict__ W,
    const float* __restrict__ bias, bf16_t* __restrict__ P,
    int K, int Lseq, int transpose)
{
  const int lane = threadIdx.x & 31;
  const int wave = threadIdx.x >> 5;
  const int lm = lane & 15, lh = lane >> 4;
  const int M0 = blockIdx.y * 128 + (wave & 3) * 32;
  const int F0 = blockIdx.x * 128 + (wave >> 2) * 64;
  const int E  = K;

  v8f acc[2][4] = {};
  const float* arow0 = X + (size_t)(M0 + lm) * K;
  const float* arow1 = X + (size_t)(M0 + 16 + lm) * K;
  const float* brow[4];
#pragma unroll
  for (int nt = 0; nt < 4; ++nt) brow[nt] = W + (size_t)(F0 + nt * 16 + lm) * K;
  const int offA = lh * 16, offB = lh * 16;

  for (int k0 = 0; k0 < K; k0 += 32) {
    if (k0 + 64 < K) {  // lowers to global_prefetch_b8
      __builtin_prefetch(arow0 + k0 + 64, 0, 0);
      __builtin_prefetch(arow1 + k0 + 64, 0, 0);
    }
    v16bf a0 = loadA_f32(arow0 + k0 + offA);
    v16bf a1 = loadA_f32(arow1 + k0 + offA);
#pragma unroll
    for (int nt = 0; nt < 4; ++nt) {
      v16bf b = loadB_f32(brow[nt] + k0 + offB);
      acc[0][nt] = wmma_bf16(a0, b, acc[0][nt]);
      acc[1][nt] = wmma_bf16(a1, b, acc[1][nt]);
    }
  }

#pragma unroll
  for (int mt = 0; mt < 2; ++mt)
#pragma unroll
    for (int nt = 0; nt < 4; ++nt) {
      const int f = F0 + nt * 16 + lm;
      const float bv = bias[f];
      const int mbase = M0 + mt * 16 + lh * 8;
      if (transpose) {  // stored (n, E, Lseq): contiguous along l per lane
        const int n  = mbase / Lseq;
        const int l0 = mbase - n * Lseq;
        v8bf pk;
#pragma unroll
        for (int i = 0; i < 8; ++i) pk[i] = (__bf16)(acc[mt][nt][i] + bv);
        *(v8bf*)(P + ((size_t)n * E + f) * (size_t)Lseq + l0) = pk;
      } else {
#pragma unroll
        for (int i = 0; i < 8; ++i)
          P[(size_t)(mbase + i) * E + f] = (__bf16)(acc[mt][nt][i] + bv);
      }
    }
}

// ========= LDS-staged GEMM: C[M,N] = A(MxK) * B(NxK)^T, bf16 in, f32 acc =========
// Block 128x128 (8 waves of 32x64). TDM double-buffered staging of 128x32 slabs.
__global__ __launch_bounds__(256) void gemm_lds_kernel(
    const bf16_t* __restrict__ A, const bf16_t* __restrict__ B,
    void* __restrict__ OUT, int K, int ncols, float scale, int out_f32)
{
  __shared__ bf16_t ldsA[2][128 * ROWPAD];
  __shared__ bf16_t ldsB[2][128 * ROWPAD];

  const int tid = threadIdx.x;
  const int lane = tid & 31, wave = tid >> 5;
  const int lm = lane & 15, lh = lane >> 4;
  const int M0 = blockIdx.y * 128, N0 = blockIdx.x * 128;
  const int ml0 = (wave & 3) * 32, nl0 = (wave >> 2) * 64;
  const int nk = K >> 5;

  v8f acc[2][4] = {};

#if USE_TDM
  const int Arem = gridDim.y * 128 - M0;   // rows remaining from tile start (OOB guard)
  const int Brem = gridDim.x * 128 - N0;
  if (wave == 0) {
    tdm_load_tile(A + (size_t)M0 * K, (unsigned)(size_t)(void*)&ldsA[0][0], K, Arem, K);
    tdm_load_tile(B + (size_t)N0 * K, (unsigned)(size_t)(void*)&ldsB[0][0], K, Brem, K);
  }
  for (int ks = 0; ks < nk; ++ks) {
    const int buf = ks & 1;
    if (wave == 0) {
      if (ks + 1 < nk) {
        const int k1 = (ks + 1) << 5;
        tdm_load_tile(A + (size_t)M0 * K + k1, (unsigned)(size_t)(void*)&ldsA[buf ^ 1][0],
                      K - k1, Arem, K);
        tdm_load_tile(B + (size_t)N0 * K + k1, (unsigned)(size_t)(void*)&ldsB[buf ^ 1][0],
                      K - k1, Brem, K);
        __builtin_amdgcn_s_wait_tensorcnt((short)2);  // stage ks complete (in-order)
      } else {
        __builtin_amdgcn_s_wait_tensorcnt((short)0);
      }
    }
    __syncthreads();
    compute_tiles(&ldsA[buf][0], &ldsB[buf][0], ml0, nl0, lm, lh, acc);
    __syncthreads();
  }
#else
  for (int ks = 0; ks < nk; ++ks) {
    const int buf = ks & 1;
    const int k0 = ks << 5;
    stage_tile(A + (size_t)M0 * K + k0, &ldsA[buf][0], K, tid);
    stage_tile(B + (size_t)N0 * K + k0, &ldsB[buf][0], K, tid);
    __syncthreads();
    compute_tiles(&ldsA[buf][0], &ldsB[buf][0], ml0, nl0, lm, lh, acc);
    __syncthreads();
  }
#endif

#pragma unroll
  for (int mt = 0; mt < 2; ++mt)
#pragma unroll
    for (int nt = 0; nt < 4; ++nt) {
      const int col = N0 + nl0 + nt * 16 + lm;
      const int mbase = M0 + ml0 + mt * 16 + lh * 8;
      if (out_f32) {
        float* O = (float*)OUT;
#pragma unroll
        for (int i = 0; i < 8; ++i)
          O[(size_t)(mbase + i) * ncols + col] = acc[mt][nt][i];
      } else {
        bf16_t* O = (bf16_t*)OUT;
#pragma unroll
        for (int i = 0; i < 8; ++i)
          O[(size_t)(mbase + i) * ncols + col] = (__bf16)(acc[mt][nt][i] * scale);
      }
    }
}

// ================= softmax over rows (L=4096, 256 threads/row) =================
__global__ __launch_bounds__(256) void softmax_kernel(bf16_t* __restrict__ S, int Lr)
{
  __shared__ float redm[8];
  __shared__ float reds[8];
  const int tid = threadIdx.x;
  const int lane = tid & 31, wave = tid >> 5;
  bf16_t* row = S + (size_t)blockIdx.x * Lr;

  float v[16];
  float m = -3.0e38f;
#pragma unroll
  for (int i = 0; i < 16; ++i) {
    v[i] = (float)row[tid + i * 256];
    m = fmaxf(m, v[i]);
  }
#pragma unroll
  for (int o = 16; o > 0; o >>= 1) m = fmaxf(m, __shfl_xor(m, o, 32));
  if (lane == 0) redm[wave] = m;
  __syncthreads();
  float bm = redm[0];
#pragma unroll
  for (int i = 1; i < 8; ++i) bm = fmaxf(bm, redm[i]);

  float s = 0.f;
#pragma unroll
  for (int i = 0; i < 16; ++i) { v[i] = __expf(v[i] - bm); s += v[i]; }
#pragma unroll
  for (int o = 16; o > 0; o >>= 1) s += __shfl_xor(s, o, 32);
  if (lane == 0) reds[wave] = s;
  __syncthreads();
  float total = 0.f;
#pragma unroll
  for (int i = 0; i < 8; ++i) total += reds[i];
  const float inv = 1.0f / total;
#pragma unroll
  for (int i = 0; i < 16; ++i) row[tid + i * 256] = (__bf16)(v[i] * inv);
}

// ================= host launcher =================
extern "C" void kernel_launch(void* const* d_in, const int* in_sizes, int n_in,
                              void* d_out, int out_size, void* d_ws, size_t ws_size,
                              hipStream_t stream) {
  const int N = 4, L = 4096, E = 1024;
  const float* q  = (const float*)d_in[0];
  const float* k  = (const float*)d_in[1];
  const float* v  = (const float*)d_in[2];
  const float* Wq = (const float*)d_in[3];
  const float* bq = (const float*)d_in[4];
  const float* Wk = (const float*)d_in[5];
  const float* bk = (const float*)d_in[6];
  const float* Wv = (const float*)d_in[7];
  const float* bv = (const float*)d_in[8];
  float* out = (float*)d_out;

  const size_t PL = (size_t)N * L * E;
  bf16_t* pq  = (bf16_t*)d_ws;   // (N*L, E)
  bf16_t* pk  = pq  + PL;        // (N*L, E)
  bf16_t* pvT = pk  + PL;        // (N, E, L)
  bf16_t* S   = pvT + PL;        // (L, L) scratch, reused per batch

  dim3 blk(256);
  proj_kernel<<<dim3(E / 128, (N * L) / 128), blk, 0, stream>>>(q, Wq, bq, pq,  E, L, 0);
  proj_kernel<<<dim3(E / 128, (N * L) / 128), blk, 0, stream>>>(k, Wk, bk, pk,  E, L, 0);
  proj_kernel<<<dim3(E / 128, (N * L) / 128), blk, 0, stream>>>(v, Wv, bv, pvT, E, L, 1);

  const float scale = 0.03125f;  // 1/sqrt(1024)
  for (int n = 0; n < N; ++n) {
    const size_t po = (size_t)n * L * E;
    gemm_lds_kernel<<<dim3(L / 128, L / 128), blk, 0, stream>>>(
        pq + po, pk + po, (void*)S, E, L, scale, 0);
    softmax_kernel<<<dim3(L), blk, 0, stream>>>(S, L);
    gemm_lds_kernel<<<dim3(E / 128, L / 128), blk, 0, stream>>>(
        S, pvT + (size_t)n * E * L, (void*)(out + po), L, E, 1.0f, 1);
  }
}
